// RingAttention_21311627723171
// MI455X (gfx1250) — matrix operand order repaired and checked by reading, not compiled, not run on hardware
//
#include <hip/hip_runtime.h>
#include <hip/hip_bf16.h>

typedef __bf16 bf16_t;
typedef __attribute__((ext_vector_type(16))) __bf16 v16bf;
typedef __attribute__((ext_vector_type(8)))  __bf16 v8bf;
typedef __attribute__((ext_vector_type(8)))  float  v8f;
typedef __attribute__((ext_vector_type(4))) unsigned int u32x4;
typedef __attribute__((ext_vector_type(8))) int          i32x8;
typedef __attribute__((ext_vector_type(4))) int          i32x4;

#define DIM     1024
#define SEQ     4096
#define HEADS   8
#define DHEAD   64
#define DINNER  512
#define NQKV    1536

#if defined(__has_builtin)
#if __has_builtin(__builtin_amdgcn_tensor_load_to_lds) && \
    __has_builtin(__builtin_amdgcn_s_wait_tensorcnt)
#define HAVE_TDM 1
#endif
#endif
#ifndef HAVE_TDM
#define HAVE_TDM 0
#endif

// ---------------- Tensor Data Mover: 2D tile Global->LDS --------------------
// Descriptor per CDNA5 ISA (D# groups 0/1; groups 2/3 zero).
#if HAVE_TDM
__device__ __forceinline__ void tdm_load_2d(unsigned lds_off, const void* gptr,
                                            unsigned tile_w, unsigned tile_h,
                                            unsigned row_stride_elems) {
    unsigned long long ga = (unsigned long long)gptr;
    u32x4 g0;
    g0[0] = 1u;                                        // count=1, user desc
    g0[1] = lds_off;                                   // lds_addr (bytes)
    g0[2] = (unsigned)ga;                              // global_addr[31:0]
    g0[3] = (unsigned)(ga >> 32) | (2u << 30);         // global_addr[56:32] | type=2
    i32x8 g1;
    g1[0] = (int)(1u << 16);                           // data_size=1 (2 bytes/elem)
    g1[1] = (int)((tile_w & 0xFFFFu) << 16);           // tensor_dim0[15:0] @bit48
    g1[2] = (int)((tile_h & 0xFFFFu) << 16);           // dim0 hi=0 | tensor_dim1[15:0]
    g1[3] = (int)((tile_w & 0xFFFFu) << 16);           // dim1 hi=0 | tile_dim0
    g1[4] = (int)(tile_h & 0xFFFFu);                   // tile_dim1 | tile_dim2=0
    g1[5] = (int)row_stride_elems;                     // tensor_dim0_stride[31:0]
    g1[6] = 0;
    g1[7] = 0;
    i32x4 z4 = {0, 0, 0, 0};
    i32x8 z8 = {0, 0, 0, 0, 0, 0, 0, 0};
    __builtin_amdgcn_tensor_load_to_lds(g0, g1, z4, z4, z8, 0);
}
__device__ __forceinline__ void tdm_wait() { __builtin_amdgcn_s_wait_tensorcnt(0); }
__device__ __forceinline__ unsigned lds_off_of(const void* p) {
    return (unsigned)(unsigned long long)p;            // flat->LDS: addr[31:0]
}
#endif

// ---------------- WMMA fragment helpers (layouts per CDNA5 ISA) ---

// A-matrix 16x32 bf16: lane m = lane%16, half = lane/16.
// VGPR0-3 <- K = half*8 + 0..7, VGPR4-7 <- K = 16 + half*8 + 0..7.
__device__ __forceinline__ v16bf load_frag_a(const bf16_t* p, int stride) {
    int lane = threadIdx.x & 31;
    int m = lane & 15, half = lane >> 4;
    const bf16_t* r = p + m * stride;
    union { v16bf v; v8bf h[2]; } u;
    u.h[0] = *(const v8bf*)(r + half * 8);
    u.h[1] = *(const v8bf*)(r + 16 + half * 8);
    return u.v;
}

// B-matrix 32x16 bf16: lane col c = lane%16, kh = lane/16.
// VGPR0-3 <- K = kh*8 + 0..7, VGPR4-7 <- K = 16 + kh*8 + 0..7 (matches A).
__device__ __forceinline__ v16bf load_frag_b(const bf16_t* p, int stride) {
    int lane = threadIdx.x & 31;
    int c = lane & 15, kh = lane >> 4;
    const bf16_t* r = p + c * stride + kh * 8;
    union { v16bf v; v8bf h[2]; } u;
    u.h[0] = *(const v8bf*)(r);
    u.h[1] = *(const v8bf*)(r + 16);
    return u.v;
}

__device__ __forceinline__ v8f wmma_bf16(v16bf a, v16bf b, v8f c) {
    return __builtin_amdgcn_wmma_f32_16x16x32_bf16(false, a, false, b,
                                                   (short)0, c, false, false);
}

// ---------------- elementwise / rmsnorm ------------------------------------

__global__ void k_f32_to_bf16(const float* __restrict__ in,
                              bf16_t* __restrict__ out, int n) {
    int i = blockIdx.x * blockDim.x + threadIdx.x;
    if (i < n) out[i] = (bf16_t)in[i];
}

__global__ void k_rmsnorm(const float* __restrict__ x,
                          const float* __restrict__ gamma,
                          bf16_t* __restrict__ out) {
    __shared__ float red[256];
    int row = blockIdx.x;
    const float* xr = x + row * DIM;
    float s = 0.f;
    for (int i = threadIdx.x; i < DIM; i += 256) { float v = xr[i]; s += v * v; }
    red[threadIdx.x] = s;
    __syncthreads();
    for (int off = 128; off > 0; off >>= 1) {
        if (threadIdx.x < off) red[threadIdx.x] += red[threadIdx.x + off];
        __syncthreads();
    }
    float inv = 32.0f / fmaxf(sqrtf(red[0]), 1e-12f);  // sqrt(1024)=32
    for (int i = threadIdx.x; i < DIM; i += 256)
        out[row * DIM + i] = (bf16_t)(xr[i] * inv * gamma[i]);
}

// ---------------- shared GEMM core: 64x128 block tile, BK=64, 8 waves -------

__device__ __forceinline__ void gemm_core(const bf16_t* __restrict__ A,
                                          const bf16_t* __restrict__ B,
                                          int K, int N, int m0, int n0,
                                          bf16_t* As, bf16_t* Bs,
                                          v8f acc[2][2]) {
    const int tid  = threadIdx.x;
    const int wave = tid >> 5;
    const int wm = (wave >> 2) * 32;
    const int wn = (wave & 3) * 32;
#pragma unroll
    for (int i = 0; i < 2; i++)
#pragma unroll
        for (int j = 0; j < 2; j++)
#pragma unroll
            for (int e = 0; e < 8; e++) acc[i][j][e] = 0.f;

    for (int k0 = 0; k0 < K; k0 += 64) {
        // --- Stage A tile 64x64 row-major into As ---
#if HAVE_TDM
        if (wave == 0)
            tdm_load_2d(lds_off_of(As), A + (size_t)m0 * K + k0, 64, 64, K);
#else
        {
            int row = tid >> 2, colb = (tid & 3) * 16;
            const bf16_t* src = A + (size_t)(m0 + row) * K + k0 + colb;
            *(v8bf*)(As + row * 64 + colb)     = *(const v8bf*)(src);
            *(v8bf*)(As + row * 64 + colb + 8) = *(const v8bf*)(src + 8);
        }
#endif
        // --- Stage B tile 64x128, transposed into Bs[n][k] (K-contiguous) ---
#pragma unroll
        for (int c = 0; c < 4; c++) {
            int linear = tid * 4 + c;         // 0..1023
            int kr = linear >> 4;             // 0..63
            int nc = (linear & 15) * 8;       // 0..120
            v8bf v = *(const v8bf*)(B + (size_t)(k0 + kr) * N + n0 + nc);
#pragma unroll
            for (int e = 0; e < 8; e++) Bs[(nc + e) * 64 + kr] = v[e];
        }
        if (k0 + 64 < K) {   // global_prefetch_b8
            __builtin_prefetch(A + (size_t)(m0 + (tid >> 2)) * K + k0 + 64, 0, 0);
            __builtin_prefetch(B + (size_t)(k0 + 64 + (tid >> 4)) * N + n0, 0, 0);
        }
#if HAVE_TDM
        if (wave == 0) tdm_wait();
#endif
        __syncthreads();

#pragma unroll
        for (int kk = 0; kk < 64; kk += 32) {
            v16bf af0 = load_frag_a(As + (wm +  0) * 64 + kk, 64);
            v16bf af1 = load_frag_a(As + (wm + 16) * 64 + kk, 64);
            v16bf bf0 = load_frag_b(Bs + (wn +  0) * 64 + kk, 64);
            v16bf bf1 = load_frag_b(Bs + (wn + 16) * 64 + kk, 64);
            acc[0][0] = wmma_bf16(af0, bf0, acc[0][0]);
            acc[0][1] = wmma_bf16(af0, bf1, acc[0][1]);
            acc[1][0] = wmma_bf16(af1, bf0, acc[1][0]);
            acc[1][1] = wmma_bf16(af1, bf1, acc[1][1]);
        }
        __syncthreads();
    }
}

// ---------------- QKV GEMM: scatter into head-major Q/K/V (bf16) ------------

__global__ void __launch_bounds__(256)
k_qkv_gemm(const bf16_t* __restrict__ Xn, const bf16_t* __restrict__ Wqkv,
           bf16_t* __restrict__ Qg, bf16_t* __restrict__ Kg,
           bf16_t* __restrict__ Vg) {
    __shared__ bf16_t As[64 * 64];
    __shared__ bf16_t Bs[128 * 64];
    int m0 = blockIdx.y * 64;
    int n0 = blockIdx.x * 128;
    v8f acc[2][2];
    gemm_core(Xn, Wqkv, DIM, NQKV, m0, n0, As, Bs, acc);

    int lane = threadIdx.x & 31, wave = threadIdx.x >> 5;
    int wm = (wave >> 2) * 32, wn = (wave & 3) * 32;
    int cc = lane & 15, half = lane >> 4;
#pragma unroll
    for (int i = 0; i < 2; i++)
#pragma unroll
        for (int j = 0; j < 2; j++)
#pragma unroll
            for (int r = 0; r < 8; r++) {
                int grow = m0 + wm + i * 16 + half * 8 + r;
                int gcol = n0 + wn + j * 16 + cc;
                int which = gcol >> 9;        // 0:q 1:k 2:v
                int head  = (gcol >> 6) & 7;
                int dd    = gcol & 63;
                bf16_t* dst = (which == 0) ? Qg : (which == 1) ? Kg : Vg;
                dst[((size_t)(head * SEQ) + grow) * DHEAD + dd] = (bf16_t)acc[i][j][r];
            }
}

// ---------------- Flash attention: WG = (64 q-rows, 1 head), causal ---------

__global__ void __launch_bounds__(256)
k_attn(const bf16_t* __restrict__ Qg, const bf16_t* __restrict__ Kg,
       const bf16_t* __restrict__ Vg, bf16_t* __restrict__ Ao) {
    __shared__ bf16_t Qs[64 * 64];
    __shared__ bf16_t Ks[64 * 64];
    __shared__ bf16_t Vt[64 * 64];   // transposed: Vt[d][j]
    __shared__ float  Sf[64 * 64];
    __shared__ bf16_t Pb[64 * 64];
    __shared__ float  mrow[64], lrow[64], arow[64];

    const int qb = blockIdx.x, head = blockIdx.y;
    const int tid = threadIdx.x;
    const int lane = tid & 31, wave = tid >> 5;
    const int wr = (wave >> 1) * 16;  // 0,16,32,48
    const int wc = (wave & 1) * 32;   // 0,32
    const int cc = lane & 15, half = lane >> 4;
    const int q0 = qb * 64;

    const bf16_t* Qh = Qg + (size_t)head * SEQ * DHEAD;
    const bf16_t* Kh = Kg + (size_t)head * SEQ * DHEAD;
    const bf16_t* Vh = Vg + (size_t)head * SEQ * DHEAD;

    // Stage Q tile 64x64
#if HAVE_TDM
    if (wave == 0) {
        tdm_load_2d(lds_off_of(Qs), Qh + (size_t)q0 * DHEAD, 64, 64, DHEAD);
        tdm_wait();
    }
#else
#pragma unroll
    for (int c = 0; c < 2; c++) {
        int linear = tid * 2 + c;
        int row = linear >> 3, col = (linear & 7) * 8;
        *(v8bf*)(Qs + row * 64 + col) =
            *(const v8bf*)(Qh + (size_t)(q0 + row) * DHEAD + col);
    }
#endif
    if (tid < 64) { mrow[tid] = -3.0e38f; lrow[tid] = 0.f; }

    v8f oacc[2];
#pragma unroll
    for (int t = 0; t < 2; t++)
#pragma unroll
        for (int e = 0; e < 8; e++) oacc[t][e] = 0.f;

    for (int jb = 0; jb <= qb; jb++) {
        int j0 = jb * 64;
        __syncthreads();  // previous iteration done with Ks/Vt/Pb
        // Stage K tile (row-major) + V tile (transposed)
#if HAVE_TDM
        if (wave == 0)
            tdm_load_2d(lds_off_of(Ks), Kh + (size_t)j0 * DHEAD, 64, 64, DHEAD);
#endif
#pragma unroll
        for (int c = 0; c < 2; c++) {
            int linear = tid * 2 + c;
            int row = linear >> 3, col = (linear & 7) * 8;
#if !HAVE_TDM
            *(v8bf*)(Ks + row * 64 + col) =
                *(const v8bf*)(Kh + (size_t)(j0 + row) * DHEAD + col);
#endif
            v8bf vv = *(const v8bf*)(Vh + (size_t)(j0 + row) * DHEAD + col);
#pragma unroll
            for (int e = 0; e < 8; e++) Vt[(col + e) * 64 + row] = vv[e];
        }
#if HAVE_TDM
        if (wave == 0) tdm_wait();
#endif
        __syncthreads();

        // S = Q K^T (B-fragment reads K rows: col=j, d contiguous)
        v8f sacc[2];
#pragma unroll
        for (int t = 0; t < 2; t++)
#pragma unroll
            for (int e = 0; e < 8; e++) sacc[t][e] = 0.f;
#pragma unroll
        for (int kk = 0; kk < 64; kk += 32) {
            v16bf aq = load_frag_a(Qs + wr * 64 + kk, 64);
#pragma unroll
            for (int t = 0; t < 2; t++) {
                v16bf bk = load_frag_b(Ks + (wc + t * 16) * 64 + kk, 64);
                sacc[t] = wmma_bf16(aq, bk, sacc[t]);
            }
        }
        // scale (faithful q*sqrt(64)) + causal mask, spill to LDS
#pragma unroll
        for (int t = 0; t < 2; t++)
#pragma unroll
            for (int r = 0; r < 8; r++) {
                int il = wr + half * 8 + r;
                int jl = wc + t * 16 + cc;
                float v = sacc[t][r] * 8.0f;
                if (j0 + jl > q0 + il) v = -3.0e38f;
                Sf[il * 64 + jl] = v;
            }
        __syncthreads();

        // online softmax, one thread per row
        if (tid < 64) {
            const float* srow = Sf + tid * 64;
            float mo = mrow[tid], mn = mo;
#pragma unroll 8
            for (int j = 0; j < 64; j++) mn = fmaxf(mn, srow[j]);
            float al = __expf(mo - mn);
            float s = 0.f;
#pragma unroll 8
            for (int j = 0; j < 64; j++) {
                float p = __expf(srow[j] - mn);
                Pb[tid * 64 + j] = (bf16_t)p;
                s += p;
            }
            mrow[tid] = mn;
            lrow[tid] = lrow[tid] * al + s;
            arow[tid] = al;
        }
        __syncthreads();

        // O = O*alpha + P V (B-fragment reads Vt rows: col=d, j contiguous)
#pragma unroll
        for (int t = 0; t < 2; t++)
#pragma unroll
            for (int r = 0; r < 8; r++)
                oacc[t][r] *= arow[wr + half * 8 + r];
#pragma unroll
        for (int kk = 0; kk < 64; kk += 32) {
            v16bf ap = load_frag_a(Pb + wr * 64 + kk, 64);
#pragma unroll
            for (int t = 0; t < 2; t++) {
                v16bf bv = load_frag_b(Vt + (wc + t * 16) * 64 + kk, 64);
                oacc[t] = wmma_bf16(ap, bv, oacc[t]);
            }
        }
    }

#pragma unroll
    for (int t = 0; t < 2; t++)
#pragma unroll
        for (int r = 0; r < 8; r++) {
            int il = wr + half * 8 + r;
            int dl = wc + t * 16 + cc;
            float v = oacc[t][r] / lrow[il];
            Ao[(size_t)(q0 + il) * DINNER + head * DHEAD + dl] = (bf16_t)v;
        }
}

// ---------------- output projection GEMM -> f32 -----------------------------

__global__ void __launch_bounds__(256)
k_out_gemm(const bf16_t* __restrict__ Aoi, const bf16_t* __restrict__ Wout,
           float* __restrict__ Out) {
    __shared__ bf16_t As[64 * 64];
    __shared__ bf16_t Bs[128 * 64];
    int m0 = blockIdx.y * 64;
    int n0 = blockIdx.x * 128;
    v8f acc[2][2];
    gemm_core(Aoi, Wout, DINNER, DIM, m0, n0, As, Bs, acc);

    int lane = threadIdx.x & 31, wave = threadIdx.x >> 5;
    int wm = (wave >> 2) * 32, wn = (wave & 3) * 32;
    int cc = lane & 15, half = lane >> 4;
#pragma unroll
    for (int i = 0; i < 2; i++)
#pragma unroll
        for (int j = 0; j < 2; j++)
#pragma unroll
            for (int r = 0; r < 8; r++) {
                int grow = m0 + wm + i * 16 + half * 8 + r;
                int gcol = n0 + wn + j * 16 + cc;
                Out[(size_t)grow * DIM + gcol] = acc[i][j][r];
            }
}

// ---------------- launcher --------------------------------------------------

extern "C" void kernel_launch(void* const* d_in, const int* in_sizes, int n_in,
                              void* d_out, int out_size, void* d_ws, size_t ws_size,
                              hipStream_t stream) {
    const float* x     = (const float*)d_in[0];
    const float* gamma = (const float*)d_in[1];
    const float* w_qkv = (const float*)d_in[2];
    const float* w_out = (const float*)d_in[3];
    float* out = (float*)d_out;

    char* ws = (char*)d_ws;
    bf16_t* wqkv_b = (bf16_t*)(ws + 0);           // 1024*1536*2 = 3 MiB
    bf16_t* wout_b = (bf16_t*)(ws + (3u  << 20)); // 512*1024*2  = 1 MiB
    bf16_t* xn     = (bf16_t*)(ws + (4u  << 20)); // 4096*1024*2 = 8 MiB
    bf16_t* Qg     = (bf16_t*)(ws + (12u << 20)); // 8*4096*64*2 = 4 MiB
    bf16_t* Kg     = (bf16_t*)(ws + (16u << 20));
    bf16_t* Vg     = (bf16_t*)(ws + (20u << 20));
    bf16_t* Ao     = (bf16_t*)(ws + (24u << 20)); // 4096*512*2  = 4 MiB

    k_f32_to_bf16<<<(DIM * NQKV + 255) / 256, 256, 0, stream>>>(w_qkv, wqkv_b, DIM * NQKV);
    k_f32_to_bf16<<<(DINNER * DIM + 255) / 256, 256, 0, stream>>>(w_out, wout_b, DINNER * DIM);
    k_rmsnorm<<<SEQ, 256, 0, stream>>>(x, gamma, xn);
    k_qkv_gemm<<<dim3(NQKV / 128, SEQ / 64), 256, 0, stream>>>(xn, wqkv_b, Qg, Kg, Vg);
    k_attn<<<dim3(SEQ / 64, HEADS), 256, 0, stream>>>(Qg, Kg, Vg, Ao);
    k_out_gemm<<<dim3(DIM / 128, SEQ / 64), 256, 0, stream>>>(Ao, wout_b, out);
}
